// Decoder_75179107549596
// MI455X (gfx1250) — matrix operand order repaired and checked by reading, not compiled, not run on hardware
//
#include <hip/hip_runtime.h>
#include <hip/hip_bf16.h>

typedef __attribute__((ext_vector_type(16))) _Float16 v16h;
typedef __attribute__((ext_vector_type(2)))  _Float16 v2h;
typedef __attribute__((ext_vector_type(8)))  float    v8f;

#define HID   256
#define TLEN  8192
#define LLEN  8192
#define KDIM  64
// softmax computed in exp2 domain: x*SCALE*log2(e); v_exp_f32 is natively 2^x
#define C2    (0.125f * 1.44269504088896340736f)

// ---------------------------------------------------------------------------
// Kernel 1: serial LSTM-cell recurrence (zero-state cell => only input path).
// One persistent workgroup (768 threads = 24 wave32). Thread t owns one gate
// row (i / g / o; f-gate is dead since c_prev==0), weights resident in VGPRs
// as packed f16; h broadcast through LDS (ds_load_b128 bank-broadcast).
// ---------------------------------------------------------------------------
__global__ __launch_bounds__(768) void recurrence_kernel(
    const float* __restrict__ enc,      // (8192, 256)
    const float* __restrict__ W_ih,     // (1024, 256)
    const float* __restrict__ b_ih,     // (1024,)
    const float* __restrict__ b_hh,     // (1024,)
    float* __restrict__ hs_out)         // (8192, 256)
{
    __shared__ float hbuf[HID];
    __shared__ float act[3 * HID];      // sig(i), tanh(g), sig(o)

    const int t = threadIdx.x;
    const int q = t >> 8;               // 0:i  1:g  2:o
    const int j = t & 255;
    const int gateRow = (q == 0) ? j : (q == 1) ? (512 + j) : (768 + j);

    const float bsum = b_ih[gateRow] + b_hh[gateRow];

    const float2* wrow = (const float2*)(W_ih + (size_t)gateRow * HID);
    v2h w[128];
#pragma unroll
    for (int k = 0; k < 128; ++k) {
        float2 f = wrow[k];
        w[k] = (v2h){ (_Float16)f.x, (_Float16)f.y };
    }

    if (t < HID) hbuf[t] = enc[(size_t)(TLEN - 1) * HID + t];   // h0 = enc[-1]
    __syncthreads();

    for (int step = 0; step < TLEN; ++step) {
        float acc = bsum;
        const float4* h4 = (const float4*)hbuf;
#pragma unroll
        for (int k = 0; k < 64; ++k) {
            float4 hv = h4[k];              // broadcast LDS read
            v2h w0 = w[2 * k], w1 = w[2 * k + 1];
            acc += (float)w0[0] * hv.x;     // v_fma_mix_f32
            acc += (float)w0[1] * hv.y;
            acc += (float)w1[0] * hv.z;
            acc += (float)w1[1] * hv.w;
        }
        float a = (q == 1) ? tanhf(acc) : 1.0f / (1.0f + __expf(-acc));
        act[q * HID + j] = a;
        __syncthreads();
        if (t < HID) {
            float c  = act[t] * act[HID + t];          // sig(i)*tanh(g)
            float hh = act[2 * HID + t] * tanhf(c);    // sig(o)*tanh(c)
            hbuf[t] = hh;
            hs_out[(size_t)step * HID + t] = hh;
        }
        __syncthreads();
    }
}

// ---------------------------------------------------------------------------
// Kernel 2: Kp = enc @ Wk^T + bk ; Qp = hs @ Wq^T + bq  -> f16 into workspace.
// 256 blocks x 64 threads. Thread t keeps W row t register-resident (f16
// packed, loaded coalesced once), then streams 64 source rows through an LDS
// broadcast buffer. No scattered global reads, no LDS bank conflicts.
// ---------------------------------------------------------------------------
__global__ __launch_bounds__(64) void project_kernel(
    const float* __restrict__ enc, const float* __restrict__ hs,
    const float* __restrict__ Wk,  const float* __restrict__ bk,
    const float* __restrict__ Wq,  const float* __restrict__ bq,
    _Float16* __restrict__ Kp, _Float16* __restrict__ Qp)
{
    __shared__ float row[HID];
    const int b    = blockIdx.x;               // 0..255
    const bool isQ = (b >= 128);
    const int rows0 = (b & 127) * 64;          // 64 rows per block
    const float* src  = isQ ? hs : enc;
    const float* W    = isQ ? Wq : Wk;
    const float* bias = isQ ? bq : bk;
    _Float16*   dst   = isQ ? Qp : Kp;
    const int t = threadIdx.x;

    const float2* wrow = (const float2*)(W + (size_t)t * HID);
    v2h w[128];
#pragma unroll
    for (int k = 0; k < 128; ++k) {
        float2 f = wrow[k];
        w[k] = (v2h){ (_Float16)f.x, (_Float16)f.y };
    }
    const float bb = bias[t];

    for (int rr = 0; rr < 64; ++rr) {
        const float* s = src + (size_t)(rows0 + rr) * HID;
#pragma unroll
        for (int k = 0; k < HID; k += 64) row[k + t] = s[k + t];
        __syncthreads();
        float acc = bb;
        const float4* h4 = (const float4*)row;
#pragma unroll
        for (int k = 0; k < 64; ++k) {
            float4 hv = h4[k];
            v2h w0 = w[2 * k], w1 = w[2 * k + 1];
            acc += (float)w0[0] * hv.x + (float)w0[1] * hv.y
                 + (float)w1[0] * hv.z + (float)w1[1] * hv.w;
        }
        dst[(size_t)(rows0 + rr) * KDIM + t] = (_Float16)acc;
        __syncthreads();
    }
}

// ---------------------------------------------------------------------------
// Kernel 3: scores = Qp @ Kp^T * SCALE, fused row softmax (flash two-pass).
// One wave handles WTILES=2 row-tiles (32 rows). B tiles double-buffered,
// shared by both row-tiles (4 WMMAs per B load). Branchless exp2-domain
// online (max,sum); 4-step shfl_xor merge across the 16 lanes per row.
// ---------------------------------------------------------------------------
#define WTILES 2
#define NTILE  (LLEN / 16)

__device__ __forceinline__ void load_btile(const _Float16* __restrict__ Kp,
                                           int col0, int nl, int mg,
                                           v16h& b0, v16h& b1)
{
    // B 32x16 f16 layout: lane holds column nl, k = 8*mg + j (j<8), 16 + 8*mg + (j-8) (j>=8)
    const _Float16* krow = Kp + (size_t)(col0 + nl) * KDIM + mg * 8;
#pragma unroll
    for (int jj = 0; jj < 16; ++jj) { b0[jj] = krow[jj + (jj & 8)]; b1[jj] = krow[32 + jj + (jj & 8)]; }
}

__device__ __forceinline__ v8f wmma2(v16h a0, v16h a1, v16h b0, v16h b1)
{
    v8f c = {};
    c = __builtin_amdgcn_wmma_f32_16x16x32_f16(false, a0, false, b0, (short)0, c, false, false);
    c = __builtin_amdgcn_wmma_f32_16x16x32_f16(false, a1, false, b1, (short)0, c, false, false);
    return c;
}

__global__ __launch_bounds__(256) void scores_softmax_kernel(
    const _Float16* __restrict__ Qp,   // (8192, 64) f16
    const _Float16* __restrict__ Kp,   // (8192, 64) f16
    float* __restrict__ out)           // (8192, 8192) f32
{
    const int lane  = threadIdx.x & 31;
    const int wave  = threadIdx.x >> 5;
    const int group = blockIdx.x * 8 + wave;   // 32 blocks * 8 waves = 256
    const int nl = lane & 15;
    const int mg = lane >> 4;

    // A fragments for WTILES row-tiles (ISA 16-bit A 16x32 layout).
    v16h a0[WTILES], a1[WTILES];
#pragma unroll
    for (int wti = 0; wti < WTILES; ++wti) {
        const int row0 = (group * WTILES + wti) * 16;
        const _Float16* qrow = Qp + (size_t)(row0 + nl) * KDIM;
#pragma unroll
        for (int jj = 0; jj < 8; ++jj) {
            a0[wti][jj]     = qrow[mg * 8 + jj];
            a0[wti][jj + 8] = qrow[16 + mg * 8 + jj];
            a1[wti][jj]     = qrow[32 + mg * 8 + jj];
            a1[wti][jj + 8] = qrow[48 + mg * 8 + jj];
        }
    }

    float mrow[WTILES][8], srow[WTILES][8];
#pragma unroll
    for (int wti = 0; wti < WTILES; ++wti)
#pragma unroll
        for (int r = 0; r < 8; ++r) { mrow[wti][r] = -INFINITY; srow[wti][r] = 0.0f; }

    // ---- Pass 1: branchless online (max, sumexp2), double-buffered B ----
    v16h b0, b1, nb0, nb1;
    load_btile(Kp, 0, nl, mg, nb0, nb1);
    for (int tile = 0; tile < NTILE; ++tile) {
        b0 = nb0; b1 = nb1;
        if (tile + 1 < NTILE) load_btile(Kp, (tile + 1) * 16, nl, mg, nb0, nb1);
#pragma unroll
        for (int wti = 0; wti < WTILES; ++wti) {
            v8f c = wmma2(a0[wti], a1[wti], b0, b1);
#pragma unroll
            for (int r = 0; r < 8; ++r) {
                float x  = c[r] * C2;
                float nm = fmaxf(mrow[wti][r], x);
                srow[wti][r] = srow[wti][r] * exp2f(mrow[wti][r] - nm) + exp2f(x - nm);
                mrow[wti][r] = nm;
            }
        }
    }

    // ---- Merge the 16 lanes sharing each row ----
#pragma unroll
    for (int mask = 1; mask < 16; mask <<= 1) {
#pragma unroll
        for (int wti = 0; wti < WTILES; ++wti)
#pragma unroll
            for (int r = 0; r < 8; ++r) {
                float om = __shfl_xor(mrow[wti][r], mask, 32);
                float os = __shfl_xor(srow[wti][r], mask, 32);
                float nm = fmaxf(mrow[wti][r], om);
                srow[wti][r] = srow[wti][r] * exp2f(mrow[wti][r] - nm)
                             + os * exp2f(om - nm);
                mrow[wti][r] = nm;
            }
    }
    float inv[WTILES][8];
#pragma unroll
    for (int wti = 0; wti < WTILES; ++wti)
#pragma unroll
        for (int r = 0; r < 8; ++r) inv[wti][r] = 1.0f / srow[wti][r];

    // ---- Pass 2: recompute tiles, write normalized softmax to HBM ----
    load_btile(Kp, 0, nl, mg, nb0, nb1);
    for (int tile = 0; tile < NTILE; ++tile) {
        const int col0 = tile * 16;
        b0 = nb0; b1 = nb1;
        if (tile + 1 < NTILE) load_btile(Kp, (tile + 1) * 16, nl, mg, nb0, nb1);
#pragma unroll
        for (int wti = 0; wti < WTILES; ++wti) {
            v8f c = wmma2(a0[wti], a1[wti], b0, b1);
            const size_t rowbase = (size_t)((group * WTILES + wti) * 16 + 8 * mg);
#pragma unroll
            for (int r = 0; r < 8; ++r) {
                out[(rowbase + r) * LLEN + col0 + nl] =
                    exp2f(c[r] * C2 - mrow[wti][r]) * inv[wti][r];
            }
        }
    }
}

// ---------------------------------------------------------------------------
extern "C" void kernel_launch(void* const* d_in, const int* in_sizes, int n_in,
                              void* d_out, int out_size, void* d_ws, size_t ws_size,
                              hipStream_t stream) {
    (void)in_sizes; (void)n_in; (void)out_size; (void)ws_size;
    const float* enc  = (const float*)d_in[1];
    const float* W_ih = (const float*)d_in[2];
    const float* b_ih = (const float*)d_in[4];
    const float* b_hh = (const float*)d_in[5];
    const float* Wq   = (const float*)d_in[6];
    const float* bq   = (const float*)d_in[7];
    const float* Wk   = (const float*)d_in[8];
    const float* bk   = (const float*)d_in[9];

    float* out = (float*)d_out;
    float* hs  = out + (size_t)TLEN * LLEN;            // hs region of d_out

    _Float16* Kp = (_Float16*)d_ws;                    // 8192*64 f16 = 1 MB
    _Float16* Qp = Kp + (size_t)LLEN * KDIM;           // next 1 MB

    recurrence_kernel<<<1, 768, 0, stream>>>(enc, W_ih, b_ih, b_hh, hs);
    project_kernel<<<256, 64, 0, stream>>>(enc, hs, Wk, bk, Wq, bq, Kp, Qp);
    scores_softmax_kernel<<<32, 256, 0, stream>>>(Qp, Kp, out);
}
